// MoEFFN_23751169146915
// MI455X (gfx1250) — matrix operand order repaired and checked by reading, not compiled, not run on hardware
//
#include <hip/hip_runtime.h>
#include <cstdint>

// ---------------- problem constants ----------------
#define BT_  32768      // B*T tokens
#define DD_  768        // model dim
#define HH_  3072       // hidden dim
#define EE_  4          // experts
#define KK_  2          // top-k

typedef __attribute__((ext_vector_type(16))) __bf16       v16bf;
typedef __attribute__((ext_vector_type(8)))  float        v8f;
typedef __attribute__((ext_vector_type(4)))  unsigned int v4u;   // trivial 16B vector (HIP uint4 has ctors -> unusable in unions)

union ABu { v16bf v; v4u q[2]; unsigned short u[16]; };

__device__ __forceinline__ unsigned short f2bf(float f) {
    union { float f; unsigned u; } v; v.f = f;
    unsigned r = v.u + 0x7FFFu + ((v.u >> 16) & 1u);   // round-to-nearest-even
    return (unsigned short)(r >> 16);
}

__device__ __forceinline__ v8f wmma_bf16(v16bf a, v16bf b, v8f c) {
    // D = A(16x32 bf16) * B(32x16 bf16) + C(16x16 f32)
    return __builtin_amdgcn_wmma_f32_16x16x32_bf16(false, a, false, b, (short)0, c, false, false);
}

// ---------------- init / zero ----------------
__global__ void moe_init_kernel(int* cnt, float* probsum) {
    int t = threadIdx.x;
    if (t < EE_) { cnt[t] = 0; probsum[t] = 0.0f; }
}

__global__ void moe_zero_kernel(float* p, long long n) {
    long long i = (long long)blockIdx.x * blockDim.x + threadIdx.x;
    long long stride = (long long)gridDim.x * blockDim.x;
    for (; i < n; i += stride) p[i] = 0.0f;
}

// ---------------- weight convert/transpose to bf16 ----------------
// W1bf layout [e][h][d]  (so B-matrix column h is contiguous in d)
// W2bf layout [e][d][h]  (so B-matrix column d is contiguous in h)
__global__ void moe_prep_kernel(const float* __restrict__ W1, const float* __restrict__ W2,
                                unsigned short* __restrict__ W1bf, unsigned short* __restrict__ W2bf) {
    size_t i = (size_t)blockIdx.x * blockDim.x + threadIdx.x;
    const size_t total = (size_t)EE_ * DD_ * HH_;
    if (i >= total) return;
    {
        size_t e = i / ((size_t)DD_ * HH_);
        size_t r = i % ((size_t)DD_ * HH_);
        size_t h = r / DD_, d = r % DD_;
        W1bf[i] = f2bf(W1[(e * DD_ + d) * HH_ + h]);   // W1 is [E][D][H]
    }
    {
        size_t e = i / ((size_t)DD_ * HH_);
        size_t r = i % ((size_t)DD_ * HH_);
        size_t d = r / HH_, h = r % HH_;
        W2bf[i] = f2bf(W2[(e * HH_ + h) * DD_ + d]);   // W2 is [E][H][D]
    }
}

// ---------------- router: softmax + top-2 gather ----------------
__global__ void moe_router_kernel(const float* __restrict__ x, const float* __restrict__ rw,
                                  const float* __restrict__ rb,
                                  int* __restrict__ cnt, float* __restrict__ probsum,
                                  int* __restrict__ lists, float* __restrict__ wlists) {
    __shared__ float sp[EE_];
    int tid = threadIdx.x;
    if (tid < EE_) sp[tid] = 0.0f;
    __syncthreads();

    int wave = tid >> 5, lane = tid & 31;
    int token = blockIdx.x * 8 + wave;

    const float* xr = x + (size_t)token * DD_;
    float a0 = 0.f, a1 = 0.f, a2 = 0.f, a3 = 0.f;
    for (int i = 0; i < DD_ / 32; ++i) {
        int d = lane + 32 * i;
        float v = xr[d];
        const float* w = rw + (size_t)d * EE_;
        a0 += v * w[0]; a1 += v * w[1]; a2 += v * w[2]; a3 += v * w[3];
    }
    #pragma unroll
    for (int off = 16; off > 0; off >>= 1) {
        a0 += __shfl_xor(a0, off);
        a1 += __shfl_xor(a1, off);
        a2 += __shfl_xor(a2, off);
        a3 += __shfl_xor(a3, off);
    }
    if (lane == 0) {
        float l[EE_] = { a0 + rb[0], a1 + rb[1], a2 + rb[2], a3 + rb[3] };
        float m = l[0];
        #pragma unroll
        for (int e = 1; e < EE_; ++e) m = fmaxf(m, l[e]);
        float p[EE_], s = 0.f;
        #pragma unroll
        for (int e = 0; e < EE_; ++e) { p[e] = __expf(l[e] - m); s += p[e]; }
        float inv = 1.0f / s;
        #pragma unroll
        for (int e = 0; e < EE_; ++e) { p[e] *= inv; atomicAdd(&sp[e], p[e]); }
        int i1 = 0;
        #pragma unroll
        for (int e = 1; e < EE_; ++e) if (p[e] > p[i1]) i1 = e;
        int i2 = (i1 == 0) ? 1 : 0;
        #pragma unroll
        for (int e = 0; e < EE_; ++e) if (e != i1 && p[e] > p[i2]) i2 = e;
        int pos1 = atomicAdd(&cnt[i1], 1);
        lists[i1 * BT_ + pos1] = token;  wlists[i1 * BT_ + pos1] = p[i1];
        int pos2 = atomicAdd(&cnt[i2], 1);
        lists[i2 * BT_ + pos2] = token;  wlists[i2 * BT_ + pos2] = p[i2];
    }
    __syncthreads();
    if (tid < EE_) atomicAdd(&probsum[tid], sp[tid]);
}

// ---------------- load-balance loss (scalar) ----------------
__global__ void moe_loss_kernel(const int* __restrict__ cnt, const float* __restrict__ probsum,
                                float* __restrict__ out) {
    float c[EE_], cs = 0.f;
    #pragma unroll
    for (int e = 0; e < EE_; ++e) { c[e] = (float)cnt[e]; cs += c[e]; }
    float l = 0.f;
    #pragma unroll
    for (int e = 0; e < EE_; ++e) l += (probsum[e] / (float)BT_) * (c[e] / cs);
    out[(size_t)BT_ * DD_] = (float)EE_ * l;
}

// ---------------- main fused expert FFN (bf16 WMMA) ----------------
// grid: (BT/32 tiles, E experts), 256 threads = 8 waves.
// Per block: 32 gathered tokens. Loop H in 64-chunks:
//   GEMM1 [32x768]x[768x64] -> GELU -> bf16 LDS -> GEMM2 [32x64]x[64x768] accum in regs.
// Wave w: mtile = w>>2 (16-token row tile), ngroup = w&3 (192-col output slice / h-subtile).
__global__ void __launch_bounds__(256)
moe_ffn_kernel(const float* __restrict__ x, const float* __restrict__ b1,
               const float* __restrict__ b2,
               const unsigned short* __restrict__ W1bf, const unsigned short* __restrict__ W2bf,
               const int* __restrict__ cnt, const int* __restrict__ lists,
               const float* __restrict__ wlists, float* __restrict__ out) {
    const int e    = blockIdx.y;
    const int tile = blockIdx.x;
    const int n    = cnt[e];
    if (tile * 32 >= n) return;

    __shared__ unsigned short xs[32 * 784];   // 50176 B, bf16 x tile, row stride 784 (16B aligned)
    __shared__ unsigned short hs[32 * 80];    // 5120 B,  bf16 h chunk, row stride 80
    __shared__ int   stok[32];
    __shared__ float swgt[32];

    const int tid = threadIdx.x;
    if (tid < 32) {
        int idx = tile * 32 + tid;
        if (idx < n) { stok[tid] = lists[e * BT_ + idx]; swgt[tid] = wlists[e * BT_ + idx]; }
        else         { stok[tid] = lists[e * BT_];       swgt[tid] = 0.0f; }     // pad row, zero weight
    }
    __syncthreads();

    // stage gathered x rows into LDS as bf16: 8 threads per row, 96 floats each
    {
        int row = tid >> 3, seg = tid & 7;
        const float* src = x + (size_t)stok[row] * DD_ + seg * 96;
        unsigned short* dst = xs + row * 784 + seg * 96;
        for (int i = 0; i < 96; i += 4) {
            float4 f = *(const float4*)(src + i);
            dst[i + 0] = f2bf(f.x); dst[i + 1] = f2bf(f.y);
            dst[i + 2] = f2bf(f.z); dst[i + 3] = f2bf(f.w);
        }
    }
    __syncthreads();

    const int wave = tid >> 5, lane = tid & 31;
    const int mtile = wave >> 2, ngroup = wave & 3;
    const int hi = lane >> 4, lo = lane & 15;

    v8f acc2[12];
    v8f zero8 = {0.f, 0.f, 0.f, 0.f, 0.f, 0.f, 0.f, 0.f};
    #pragma unroll
    for (int i = 0; i < 12; ++i) acc2[i] = zero8;

    for (int hc = 0; hc < HH_; hc += 64) {
        // ---- GEMM1: h-tile [16 tok x 16 h], K over 768 ----
        v8f acc1 = zero8;
        const int h0 = hc + ngroup * 16;
        const unsigned short* wbase = W1bf + ((size_t)e * HH_ + h0 + lo) * DD_ + hi * 8;
        const unsigned short* abase = xs + (mtile * 16 + lo) * 784;
        for (int k = 0; k < DD_; k += 32) {
            ABu a, b;
            const unsigned short* ap = abase + k + hi * 8;      // A layout: two 8-half chunks
            a.q[0] = *(const v4u*)(ap);
            a.q[1] = *(const v4u*)(ap + 16);
            const v4u* bp = (const v4u*)(wbase + k);            // B layout: two 8-half chunks
            b.q[0] = bp[0]; b.q[1] = bp[2];
            acc1 = wmma_bf16(a.v, b.v, acc1);
        }
        // bias + exact-erf GELU -> bf16 h chunk in LDS
        {
            float bias1 = b1[(size_t)e * HH_ + h0 + lo];
            #pragma unroll
            for (int r = 0; r < 8; ++r) {
                int m = r + 8 * hi;                              // C layout: M = r + 8*(lane>=16)
                float v = acc1[r] + bias1;
                float g = 0.5f * v * (1.0f + erff(v * 0.70710678118654752f));
                hs[(mtile * 16 + m) * 80 + ngroup * 16 + lo] = f2bf(g);
            }
        }
        __syncthreads();

        // ---- GEMM2: accumulate y[32 x 192-slice] over this 64-h chunk ----
        ABu a0, a1;
        {
            const unsigned short* ap = hs + (mtile * 16 + lo) * 80;
            a0.q[0] = *(const v4u*)(ap + 0  + hi * 8);
            a0.q[1] = *(const v4u*)(ap + 16 + hi * 8);
            a1.q[0] = *(const v4u*)(ap + 32 + hi * 8);
            a1.q[1] = *(const v4u*)(ap + 48 + hi * 8);
        }
        #pragma unroll
        for (int nt = 0; nt < 12; ++nt) {
            int d0 = ngroup * 192 + nt * 16;
            const unsigned short* w2base = W2bf + ((size_t)e * DD_ + d0 + lo) * HH_ + hc + hi * 8;
            ABu b0, b1v;
            const v4u* bp0 = (const v4u*)(w2base);
            b0.q[0]  = bp0[0]; b0.q[1]  = bp0[2];
            const v4u* bp1 = (const v4u*)(w2base + 32);
            b1v.q[0] = bp1[0]; b1v.q[1] = bp1[2];
            acc2[nt] = wmma_bf16(a0.v, b0.v,  acc2[nt]);
            acc2[nt] = wmma_bf16(a1.v, b1v.v, acc2[nt]);
        }
        __syncthreads();
    }

    // ---- epilogue: weighted scatter-add into out ----
    #pragma unroll
    for (int nt = 0; nt < 12; ++nt) {
        int d0 = ngroup * 192 + nt * 16 + lo;
        float bias2 = b2[(size_t)e * DD_ + d0];
        #pragma unroll
        for (int r = 0; r < 8; ++r) {
            int m = mtile * 16 + r + 8 * hi;
            float wgt = swgt[m];
            if (wgt != 0.0f) {
                float val = wgt * (acc2[nt][r] + bias2);
                atomicAdd(out + (size_t)stok[m] * DD_ + d0, val);
            }
        }
    }
}

// ---------------- launch ----------------
extern "C" void kernel_launch(void* const* d_in, const int* in_sizes, int n_in,
                              void* d_out, int out_size, void* d_ws, size_t ws_size,
                              hipStream_t stream) {
    const float* x  = (const float*)d_in[0];
    const float* rw = (const float*)d_in[1];
    const float* rb = (const float*)d_in[2];
    const float* W1 = (const float*)d_in[3];
    const float* b1 = (const float*)d_in[4];
    const float* W2 = (const float*)d_in[5];
    const float* b2 = (const float*)d_in[6];
    float* out = (float*)d_out;

    // workspace layout (all offsets 16B aligned)
    char* ws = (char*)d_ws;
    int*   cnt     = (int*)(ws);                       // 16 B
    float* probsum = (float*)(ws + 64);                // 16 B
    int*   lists   = (int*)(ws + 128);                 // E*BT*4  = 512 KB
    float* wlists  = (float*)(ws + 128 + (size_t)EE_ * BT_ * 4);           // 512 KB
    unsigned short* W1bf = (unsigned short*)(ws + 128 + 2 * (size_t)EE_ * BT_ * 4);
    unsigned short* W2bf = W1bf + (size_t)EE_ * DD_ * HH_;                 // 2 x 18.9 MB

    moe_init_kernel<<<1, 32, 0, stream>>>(cnt, probsum);
    moe_zero_kernel<<<8192, 256, 0, stream>>>(out, (long long)out_size);

    {
        size_t total = (size_t)EE_ * DD_ * HH_;
        int grid = (int)((total + 255) / 256);
        moe_prep_kernel<<<grid, 256, 0, stream>>>(W1, W2, W1bf, W2bf);
    }

    moe_router_kernel<<<BT_ / 8, 256, 0, stream>>>(x, rw, rb, cnt, probsum, lists, wlists);
    moe_loss_kernel<<<1, 1, 0, stream>>>(cnt, probsum, out);

    dim3 grid(BT_ / 32, EE_);   // 1024 token tiles (upper bound) x 4 experts; empty tiles early-exit
    moe_ffn_kernel<<<grid, 256, 0, stream>>>(x, b1, b2, W1bf, W2bf, cnt, lists, wlists, out);
}
